// ITERMEM_84920093377196
// MI455X (gfx1250) — matrix operand order repaired and checked by reading, not compiled, not run on hardware
//
#include <hip/hip_runtime.h>
#include <math.h>

// ---------------------------------------------------------------------------
// Model constants (from the reference)
// ---------------------------------------------------------------------------
#define G_     64
#define S_     32
#define P_     8
#define D_IN_  768
#define D_H_   256
#define HEADS_ 4
#define D_FF_  2048
#define NH_T_  4
#define HD_    64      // transformer head dim
#define B_     (G_*P_) // 512 sequences

typedef __attribute__((ext_vector_type(16))) _Float16 v16h;
typedef __attribute__((ext_vector_type(8)))  _Float16 v8h;
typedef __attribute__((ext_vector_type(8)))  float    v8f;

__device__ __forceinline__ v8h ld8h(const _Float16* p) { return *(const v8h*)p; }

__device__ __forceinline__ v16h cat16(v8h lo, v8h hi) {
  v16h f;
#pragma unroll
  for (int i = 0; i < 8; i++) { f[i] = lo[i]; f[8 + i] = hi[i]; }
  return f;
}

// ---------------------------------------------------------------------------
// WMMA fragment layouts (ISA 7.12.2), fragments loaded from f16 memory:
// A fragment (16x32, M x K):
//   lanes 0-15 : M=lane,     elems 0..7 -> K = k0+0..7,   elems 8..15 -> K = k0+16..23
//   lanes 16-31: M=lane-16,  elems 0..7 -> K = k0+8..15,  elems 8..15 -> K = k0+24..31
//   -> per lane: 8 halves at (row,k0+hl*8) and 8 halves at (row,k0+16+hl*8)
// B fragment (32x16, K x N) from W (N,K) row-major (== B column-major):
//   lanes 0-15 : N=lane,     elems 0..7 -> K = k0+0..7,   elems 8..15 -> K = k0+16..23
//   lanes 16-31: N=lane-16,  elems 0..7 -> K = k0+8..15,  elems 8..15 -> K = k0+24..31
//   -> per lane: 8 halves at (row,k0+hl*8) and 8 halves at (row,k0+16+hl*8)
// Rows are pre-clamped; out-of-range lanes are masked at the store.
// ---------------------------------------------------------------------------

// ---------------------------------------------------------------------------
// Generic GEMM: C[M,N] = act( A[M,K] * W[N,K]^T + bias ) (+= C if acc flag)
// A and W are f16; accumulation fp32. flags: bit0 relu, bit1 acc (f32 C),
// bit2 store f16 to C16 instead of f32 C.
// One wave32 computes a 16x64 tile: one A fragment reused across 4 B
// fragments / 4 independent accumulators; K-loop unrolled by 2 with explicit
// ping-pong fragment buffers (10 b128 loads per half-block, no conversions).
// REQUIREMENT: K multiple of 64 (all call sites: 256/768/1536).
// ---------------------------------------------------------------------------
__global__ __launch_bounds__(32) void gemm16x64(
    const _Float16* __restrict__ A, const _Float16* __restrict__ W,
    const float* __restrict__ bias, float* __restrict__ C,
    _Float16* __restrict__ C16,
    int M, int N, int K, int lda, int ldw, int ldc, int flags) {
  const int tm  = blockIdx.x * 16;
  const int tn0 = blockIdx.y * 64;
  const int lane = threadIdx.x & 31;
  const int hl = lane >> 4;
  const int r  = lane & 15;

  const int arow = min(tm + r, M - 1);
  int wrow[4];
#pragma unroll
  for (int nb = 0; nb < 4; nb++) wrow[nb] = min(tn0 + nb * 16 + r, N - 1);

  const _Float16* pa = A + (size_t)arow * lda + hl * 8;
  const _Float16* pb[4];
#pragma unroll
  for (int nb = 0; nb < 4; nb++) pb[nb] = W + (size_t)wrow[nb] * ldw + hl * 8;

  auto fa = [&](int k0) -> v16h {
    return cat16(ld8h(pa + k0), ld8h(pa + k0 + 16));
  };
  auto fb = [&](int nb, int k0) -> v16h {
    return cat16(ld8h(pb[nb] + k0), ld8h(pb[nb] + k0 + 16));
  };

  v16h a0, b0[4];   // ping
  v16h a1, b1[4];   // pong
  v8f acc[4] = {{}, {}, {}, {}};

  a0 = fa(0);
#pragma unroll
  for (int nb = 0; nb < 4; nb++) b0[nb] = fb(nb, 0);

  for (int k0 = 0; k0 < K; k0 += 64) {
    a1 = fa(k0 + 32);                               // always in range (K%64==0)
#pragma unroll
    for (int nb = 0; nb < 4; nb++) b1[nb] = fb(nb, k0 + 32);
#pragma unroll
    for (int nb = 0; nb < 4; nb++)
      acc[nb] = __builtin_amdgcn_wmma_f32_16x16x32_f16(
          false, a0, false, b0[nb], (short)0, acc[nb], false, false);

    int kn = k0 + 64;
    if (kn >= K) kn = K - 32;                       // clamped dummy, last trip
    a0 = fa(kn);
#pragma unroll
    for (int nb = 0; nb < 4; nb++) b0[nb] = fb(nb, kn);
#pragma unroll
    for (int nb = 0; nb < 4; nb++)
      acc[nb] = __builtin_amdgcn_wmma_f32_16x16x32_f16(
          false, a1, false, b1[nb], (short)0, acc[nb], false, false);
  }

#pragma unroll
  for (int nb = 0; nb < 4; nb++) {
    const int n = tn0 + nb * 16 + r;
    if (n >= N) continue;
    const float bv = (bias != nullptr) ? bias[n] : 0.0f;
#pragma unroll
    for (int v = 0; v < 8; v++) {
      const int m = tm + hl * 8 + v;
      if (m < M) {
        float val = acc[nb][v] + bv;
        if (flags & 1) val = fmaxf(val, 0.0f);
        if (flags & 4) {
          C16[(size_t)m * ldc + n] = (_Float16)val;
        } else {
          if (flags & 2) val += C[(size_t)m * ldc + n];
          C[(size_t)m * ldc + n] = val;
        }
      }
    }
  }
}

// ---------------------------------------------------------------------------
// f32 -> f16 conversion (used once per weight matrix per call)
// ---------------------------------------------------------------------------
__global__ void k_cvt(const float* __restrict__ s, _Float16* __restrict__ d, int n) {
  int i = blockIdx.x * 256 + threadIdx.x;
  if (i < n) d[i] = (_Float16)s[i];
}

// ---------------------------------------------------------------------------
// Build transformer input h: f32 (for residual) + f16 mirror (for GEMMs).
// h (B_, T, 256): t<jlen from mem (B_, 32, 256), t==jlen zero
// ---------------------------------------------------------------------------
__global__ void k_build_h(const float* __restrict__ memb, float* __restrict__ h,
                          _Float16* __restrict__ h16, int jlen, int T) {
  int idx = blockIdx.x * 256 + threadIdx.x;
  int total = B_ * T * 256;
  if (idx >= total) return;
  int o = idx & 255;
  int bt = idx >> 8;
  int t = bt % T;
  int b = bt / T;
  float v = (t < jlen) ? memb[((size_t)b * S_ + t) * 256 + o] : 0.0f;
  h[idx] = v;
  h16[idx] = (_Float16)v;
}

// xj16[(g*8+p)*768+i] = x[((g*32+j)*8+p)*768+i]   (step-j slice of the input)
__global__ void k_gather_x(const float* __restrict__ x, _Float16* __restrict__ xj16,
                           int j) {
  int idx = blockIdx.x * 256 + threadIdx.x;
  if (idx >= B_ * D_IN_) return;
  int i = idx % D_IN_;
  int gp = idx / D_IN_;
  int g = gp >> 3, p = gp & 7;
  xj16[idx] = (_Float16)x[(((size_t)g * S_ + j) * P_ + p) * D_IN_ + i];
}

// spkcat16[g, 0:768] = pooled[g*8+0], spkcat16[g, 768:1536] = x[(g*32+j)*8+0]
__global__ void k_spkcat(const float* __restrict__ pooled, const float* __restrict__ x,
                         _Float16* __restrict__ cat16, int j) {
  int idx = blockIdx.x * 256 + threadIdx.x;
  if (idx >= G_ * 2 * D_IN_) return;
  int i = idx % (2 * D_IN_);
  int g = idx / (2 * D_IN_);
  float v = (i < D_IN_) ? pooled[((size_t)g * P_) * D_IN_ + i]
                        : x[(((size_t)g * S_ + j) * P_) * D_IN_ + (i - D_IN_)];
  cat16[idx] = (_Float16)v;
}

// xj16 = pooled with row p==0 replaced by spkproj (f16 output for GAT GEMM)
__global__ void k_xj(const float* __restrict__ pooled, const float* __restrict__ spkproj,
                     _Float16* __restrict__ xj16) {
  int idx = blockIdx.x * 256 + threadIdx.x;
  if (idx >= B_ * D_IN_) return;
  int i = idx % D_IN_;
  int gp = idx / D_IN_;
  int p = gp & 7, g = gp >> 3;
  float v = (p == 0) ? spkproj[(size_t)g * D_IN_ + i] : pooled[idx];
  xj16[idx] = (_Float16)v;
}

// ---------------------------------------------------------------------------
// h = LayerNorm(h + res) * gam + bet over last dim (256); writes f32 + f16.
// One wave per row.
// ---------------------------------------------------------------------------
__global__ void k_add_ln(float* __restrict__ h, _Float16* __restrict__ h16,
                         const float* __restrict__ res,
                         const float* __restrict__ gam, const float* __restrict__ bet,
                         int rows) {
  int row = blockIdx.x * 8 + threadIdx.y;
  if (row >= rows) return;
  int lane = threadIdx.x;
  size_t base = (size_t)row * 256;
  float xv[8];
  float s = 0.0f, ss = 0.0f;
#pragma unroll
  for (int i = 0; i < 8; i++) {
    int c = lane + 32 * i;
    float v = h[base + c] + res[base + c];
    xv[i] = v; s += v; ss += v * v;
  }
#pragma unroll
  for (int off = 16; off > 0; off >>= 1) {
    s  += __shfl_xor(s,  off, 32);
    ss += __shfl_xor(ss, off, 32);
  }
  float mean = s * (1.0f / 256.0f);
  float var  = ss * (1.0f / 256.0f) - mean * mean;
  float inv  = rsqrtf(var + 1e-5f);
#pragma unroll
  for (int i = 0; i < 8; i++) {
    int c = lane + 32 * i;
    float v = (xv[i] - mean) * inv * gam[c] + bet[c];
    h[base + c] = v;
    h16[base + c] = (_Float16)v;
  }
}

// ---------------------------------------------------------------------------
// MHA core: per (sequence b, head hh, query t). T <= 33, head dim 64.
// qkv: (B_*T, 768) f32 with q|k|v at col offsets 0/256/512.
// Output written as f16 (feeds the out-projection GEMM only).
// ---------------------------------------------------------------------------
__global__ void k_attn(const float* __restrict__ qkv, _Float16* __restrict__ out16,
                       int T) {
  int idx = blockIdx.x * 128 + threadIdx.x;
  int total = B_ * NH_T_ * T;
  if (idx >= total) return;
  int t = idx % T;
  int bh = idx / T;
  int hh = bh & 3;
  int b  = bh >> 2;

  const float* q = qkv + ((size_t)b * T + t) * 768 + hh * HD_;
  float qr[HD_];
#pragma unroll
  for (int d = 0; d < HD_; d++) qr[d] = q[d];

  float sc[34];
  float m = -1e30f;
  for (int u = 0; u < T; u++) {
    const float* kp = qkv + ((size_t)b * T + u) * 768 + 256 + hh * HD_;
    float s = 0.0f;
#pragma unroll
    for (int d = 0; d < HD_; d++) s += qr[d] * kp[d];
    s *= 0.125f;  // 1/sqrt(64)
    sc[u] = s;
    m = fmaxf(m, s);
  }
  float z = 0.0f;
  for (int u = 0; u < T; u++) { float e = __expf(sc[u] - m); sc[u] = e; z += e; }
  float rz = 1.0f / z;

  float o[HD_];
#pragma unroll
  for (int d = 0; d < HD_; d++) o[d] = 0.0f;
  for (int u = 0; u < T; u++) {
    const float* vp = qkv + ((size_t)b * T + u) * 768 + 512 + hh * HD_;
    float w = sc[u] * rz;
#pragma unroll
    for (int d = 0; d < HD_; d++) o[d] += w * vp[d];
  }
  _Float16* op = out16 + ((size_t)b * T + t) * 256 + hh * HD_;
#pragma unroll
  for (int d = 0; d < HD_; d++) op[d] = (_Float16)(o[d]);
}

// ---------------------------------------------------------------------------
// GAT attention + combine. hg: (512, 1024) f32, col = head*256 + o.
// Node 0 attends only to itself; node d>=1 attends to {0, d}.
// Writes hcur (f32) into mem[(g*8+d), j, :]; f16 spk row for the classifier.
// ---------------------------------------------------------------------------
__global__ void k_gat(const float* __restrict__ hg, const float* __restrict__ as,
                      const float* __restrict__ ad, const float* __restrict__ gb,
                      float* __restrict__ memb, _Float16* __restrict__ spkb16, int j) {
  __shared__ float es_s[P_][HEADS_];
  __shared__ float ed_s[P_][HEADS_];
  int g = blockIdx.x;
  int t = threadIdx.x;  // 256

  {
    int pair = t >> 3;  // 0..31 = (p, hh)
    int sub  = t & 7;
    int p  = pair >> 2;
    int hh = pair & 3;
    const float* hrow = hg + ((size_t)g * P_ + p) * 1024 + hh * 256;
    const float* asr  = as + hh * 256;
    const float* adr  = ad + hh * 256;
    float ssum = 0.0f, dsum = 0.0f;
    for (int o = sub * 32; o < sub * 32 + 32; o++) {
      float hv = hrow[o];
      ssum += hv * asr[o];
      dsum += hv * adr[o];
    }
#pragma unroll
    for (int off = 4; off > 0; off >>= 1) {
      ssum += __shfl_down(ssum, off, 8);
      dsum += __shfl_down(dsum, off, 8);
    }
    if (sub == 0) { es_s[p][hh] = ssum; ed_s[p][hh] = dsum; }
  }
  __syncthreads();

  int o = t;  // output feature 0..255
  for (int d = 0; d < P_; d++) {
    float acc = 0.0f;
#pragma unroll
    for (int hh = 0; hh < HEADS_; hh++) {
      float h0 = hg[((size_t)g * P_ + 0) * 1024 + hh * 256 + o];
      if (d == 0) {
        acc += h0;  // only self-edge -> softmax weight 1
      } else {
        float hd = hg[((size_t)g * P_ + d) * 1024 + hh * 256 + o];
        float w0 = es_s[0][hh] + ed_s[d][hh];
        float w1 = es_s[d][hh] + ed_s[d][hh];
        w0 = (w0 > 0.0f) ? w0 : 0.2f * w0;  // leaky relu
        w1 = (w1 > 0.0f) ? w1 : 0.2f * w1;
        float mx = fmaxf(w0, w1);
        float e0 = __expf(w0 - mx), e1 = __expf(w1 - mx);
        float rzz = 1.0f / (e0 + e1);
        acc += (e0 * h0 + e1 * hd) * rzz;
      }
    }
    float val = acc * 0.25f + gb[o];  // mean over 4 heads + bias
    memb[(((size_t)g * P_ + d) * S_ + j) * 256 + o] = val;
    if (d == 0) spkb16[((size_t)g * S_ + j) * 256 + o] = (_Float16)val;
  }
}

// ---------------------------------------------------------------------------
// Host driver: sequential 32-step recurrence, all launches on `stream`.
// ---------------------------------------------------------------------------
extern "C" void kernel_launch(void* const* d_in, const int* in_sizes, int n_in,
                              void* d_out, int out_size, void* d_ws, size_t ws_size,
                              hipStream_t stream) {
  (void)in_sizes; (void)n_in; (void)out_size; (void)ws_size;

  const float* x      = (const float*)d_in[0];
  const float* gat_w  = (const float*)d_in[3];
  const float* a_s    = (const float*)d_in[4];
  const float* a_d    = (const float*)d_in[5];
  const float* gat_b  = (const float*)d_in[6];
  const float* qkv_w  = (const float*)d_in[7];
  const float* qkv_b  = (const float*)d_in[8];
  const float* out_w  = (const float*)d_in[9];
  const float* out_b  = (const float*)d_in[10];
  const float* ln1g   = (const float*)d_in[11];
  const float* ln1b   = (const float*)d_in[12];
  const float* ff1w   = (const float*)d_in[13];
  const float* ff1b   = (const float*)d_in[14];
  const float* ff2w   = (const float*)d_in[15];
  const float* ff2b   = (const float*)d_in[16];
  const float* ln2g   = (const float*)d_in[17];
  const float* ln2b   = (const float*)d_in[18];
  const float* proj_w = (const float*)d_in[19];
  const float* proj_b = (const float*)d_in[20];
  const float* spk_w  = (const float*)d_in[21];
  const float* spk_b  = (const float*)d_in[22];
  const float* cls_w  = (const float*)d_in[23];
  const float* cls_b  = (const float*)d_in[24];

  float* ws = (float*)d_ws;
  size_t off = 0;
  // f32 buffers
  float* memb = ws + off; off += (size_t)B_ * S_ * 256;   // (512, 32, 256)
  float* hbuf = ws + off; off += (size_t)B_ * 33 * 256;   // residual stream
  float* qkvb = ws + off; off += (size_t)B_ * 33 * 768;
  float* obuf = ws + off; off += (size_t)B_ * 33 * 256;
  float* f2b  = ws + off; off += (size_t)B_ * 33 * 256;
  float* pool = ws + off; off += (size_t)B_ * D_IN_;
  float* spkp = ws + off; off += (size_t)G_ * D_IN_;
  float* hgb  = ws + off; off += (size_t)B_ * 1024;
  // f16 buffers (each carved on an even-float boundary; sizes in halves)
  _Float16* h16    = (_Float16*)(ws + off); off += (size_t)B_ * 33 * 256 / 2;
  _Float16* attb16 = (_Float16*)(ws + off); off += (size_t)B_ * 33 * 256 / 2;
  _Float16* ffc16  = (_Float16*)(ws + off); off += (size_t)B_ * 33 * 256 / 2;
  _Float16* catb16 = (_Float16*)(ws + off); off += (size_t)G_ * 2 * D_IN_ / 2;
  _Float16* xjb16  = (_Float16*)(ws + off); off += (size_t)B_ * D_IN_ / 2;
  _Float16* spkb16 = (_Float16*)(ws + off); off += (size_t)G_ * S_ * 256 / 2;
  // f16 weights
  _Float16* gat_w16  = (_Float16*)(ws + off); off += (size_t)1024 * 768 / 2;
  _Float16* qkv_w16  = (_Float16*)(ws + off); off += (size_t)2 * 768 * 256 / 2;
  _Float16* out_w16  = (_Float16*)(ws + off); off += (size_t)2 * 256 * 256 / 2;
  _Float16* ff1_w16  = (_Float16*)(ws + off); off += (size_t)2 * D_FF_ * 256 / 2;
  _Float16* ff2_w16  = (_Float16*)(ws + off); off += (size_t)2 * 256 * D_FF_ / 2;
  _Float16* proj_w16 = (_Float16*)(ws + off); off += (size_t)768 * 256 / 2;
  _Float16* spk_w16  = (_Float16*)(ws + off); off += (size_t)768 * 1536 / 2;
  _Float16* cls_w16  = (_Float16*)(ws + off); off += 1024;  // 7*256 rounded up

  auto cvt = [&](const float* s, _Float16* d, int n) {
    k_cvt<<<(n + 255) / 256, 256, 0, stream>>>(s, d, n);
  };
  // one-time (per call) weight conversions
  cvt(gat_w,  gat_w16,  1024 * 768);
  cvt(qkv_w,  qkv_w16,  2 * 768 * 256);
  cvt(out_w,  out_w16,  2 * 256 * 256);
  cvt(ff1w,   ff1_w16,  2 * D_FF_ * 256);
  cvt(ff2w,   ff2_w16,  2 * 256 * D_FF_);
  cvt(proj_w, proj_w16, 768 * 256);
  cvt(spk_w,  spk_w16,  768 * 1536);
  cvt(cls_w,  cls_w16,  7 * 256);

  auto gemm = [&](const _Float16* A, const _Float16* W, const float* bias,
                  float* C, _Float16* C16,
                  int M, int N, int K, int lda, int ldw, int ldc, int flags) {
    dim3 grid((M + 15) / 16, (N + 63) / 64);
    gemm16x64<<<grid, 32, 0, stream>>>(A, W, bias, C, C16, M, N, K, lda, ldw, ldc, flags);
  };

  for (int j = 0; j < S_; j++) {
    if (j == 0) {
      k_gather_x<<<(B_ * D_IN_ + 255) / 256, 256, 0, stream>>>(x, xjb16, 0);
    } else {
      const int T = j + 1;
      const int M = B_ * T;
      k_build_h<<<(B_ * T * 256 + 255) / 256, 256, 0, stream>>>(memb, hbuf, h16, j, T);
      for (int l = 0; l < 2; l++) {
        // QKV projection
        gemm(h16, qkv_w16 + (size_t)l * 768 * 256, qkv_b + l * 768, qkvb, nullptr,
             M, 768, 256, 256, 256, 768, 0);
        // self-attention (f16 output)
        k_attn<<<(B_ * NH_T_ * T + 127) / 128, 128, 0, stream>>>(qkvb, attb16, T);
        // output projection
        gemm(attb16, out_w16 + (size_t)l * 256 * 256, out_b + l * 256, obuf, nullptr,
             M, 256, 256, 256, 256, 256, 0);
        // residual + LN1 (updates f32 stream + f16 mirror)
        k_add_ln<<<(M + 7) / 8, dim3(32, 8), 0, stream>>>(
            hbuf, h16, obuf, ln1g + l * 256, ln1b + l * 256, M);
        // FFN, chunked over ff dim (8 x 256) with fp32 accumulation in f2b
        for (int c = 0; c < 8; c++) {
          gemm(h16, ff1_w16 + (size_t)l * D_FF_ * 256 + (size_t)c * 256 * 256,
               ff1b + l * D_FF_ + c * 256, nullptr, ffc16,
               M, 256, 256, 256, 256, 256, /*relu+f16out*/ 1 | 4);
          gemm(ffc16, ff2_w16 + (size_t)l * 256 * D_FF_ + c * 256,
               (c == 0) ? (ff2b + l * 256) : nullptr, f2b, nullptr,
               M, 256, 256, 256, D_FF_, 256, (c == 0) ? 0 : /*acc*/ 2);
        }
        // residual + LN2
        k_add_ln<<<(M + 7) / 8, dim3(32, 8), 0, stream>>>(
            hbuf, h16, f2b, ln2g + l * 256, ln2b + l * 256, M);
      }
      // pooled = h[:, last] @ proj_w^T + proj_b  (strided A rows, lda = T*256)
      gemm(h16 + (size_t)j * 256, proj_w16, proj_b, pool, nullptr,
           B_, 768, 256, T * 256, 256, 768, 0);
      // speaker path
      k_spkcat<<<(G_ * 2 * D_IN_ + 255) / 256, 256, 0, stream>>>(pool, x, catb16, j);
      gemm(catb16, spk_w16, spk_b, spkp, nullptr,
           G_, 768, 2 * D_IN_, 2 * D_IN_, 2 * D_IN_, 768, /*relu*/ 1);
      k_xj<<<(B_ * D_IN_ + 255) / 256, 256, 0, stream>>>(pool, spkp, xjb16);
    }
    // GAT: projection then attention/combine (writes mem[j], spk[j])
    gemm(xjb16, gat_w16, nullptr, hgb, nullptr,
         B_, HEADS_ * D_H_, D_IN_, D_IN_, D_IN_, 1024, 0);
    k_gat<<<G_, 256, 0, stream>>>(hgb, a_s, a_d, gat_b, memb, spkb16, j);
  }

  // classifier: (2048, 256) @ (7, 256)^T + bias -> d_out (2048, 7)
  gemm(spkb16, cls_w16, cls_b, (float*)d_out, nullptr,
       G_ * S_, 7, 256, 256, 256, 7, 0);
}
